// QLinearMultiheadAttention_86457691669062
// MI455X (gfx1250) — matrix operand order repaired and checked by reading, not compiled, not run on hardware
//
#include <hip/hip_runtime.h>

// ---------------------------------------------------------------------------
// Types & helpers
// ---------------------------------------------------------------------------
typedef __attribute__((ext_vector_type(16))) __bf16          v16bf;
typedef __attribute__((ext_vector_type(8)))  float           v8f;
typedef __attribute__((ext_vector_type(8)))  unsigned short  ushort8;
typedef int v4i_gcc __attribute__((__vector_size__(16)));   // matches builtin proto

__device__ __forceinline__ unsigned short f2bf(float f) {
  unsigned int x = __builtin_bit_cast(unsigned int, f);
  x += 0x7FFFu + ((x >> 16) & 1u);          // round-to-nearest-even
  return (unsigned short)(x >> 16);
}

__device__ __forceinline__ ushort8 cvt8(float4 a, float4 b) {
  ushort8 h;
  h[0] = f2bf(a.x); h[1] = f2bf(a.y); h[2] = f2bf(a.z); h[3] = f2bf(a.w);
  h[4] = f2bf(b.x); h[5] = f2bf(b.y); h[6] = f2bf(b.z); h[7] = f2bf(b.w);
  return h;
}

// A-fragment K position within a 32-wide K block:
// lane(half=0) consumes K = 0..15 ; lane(half=1) K = 16..31 (matches B).
// permA(k) is the identity: each lane's 16 values are contiguous (32B) in LDS.
__device__ __forceinline__ int permA(int k) {
  return k;
}

// 16-byte global->LDS copy: async DMA when the toolchain exposes the gfx1250
// async-load builtins (ASYNCcnt path), else a plain b128 load/store pair.
#if defined(__has_builtin)
#if __has_builtin(__builtin_amdgcn_global_load_async_to_lds_b128) && \
    __has_builtin(__builtin_amdgcn_s_wait_asynccnt)
#define USE_ASYNC_LDS 1
#endif
#endif

__device__ __forceinline__ void cp16(unsigned short* l, const unsigned short* g) {
#if defined(USE_ASYNC_LDS)
  __builtin_amdgcn_global_load_async_to_lds_b128(
      (__attribute__((address_space(1))) v4i_gcc*)g,
      (__attribute__((address_space(3))) v4i_gcc*)l,
      0, 0);
#else
  *(ushort8*)l = *(const ushort8*)g;
#endif
}

__device__ __forceinline__ void cp_wait_barrier() {
#if defined(USE_ASYNC_LDS)
  __builtin_amdgcn_s_wait_asynccnt(0);
#endif
  __syncthreads();
}

// Load one 16-element bf16 fragment (32 bytes, 16B-aligned) from LDS.
__device__ __forceinline__ v16bf ldfrag(const unsigned short* p) {
  struct U { ushort8 lo, hi; } u;
  u.lo = *(const ushort8*)p;
  u.hi = *(const ushort8*)(p + 8);
  return __builtin_bit_cast(v16bf, u);
}

__device__ __forceinline__ v8f wmma_bf16(v16bf a, v16bf b, v8f c) {
  return __builtin_amdgcn_wmma_f32_16x16x32_bf16(
      false, a, false, b, (short)0, c, false, false);
}

#define S_LEN 2048
#define D_EMB 1024
#define KP    256
#define DK    64

// ---------------------------------------------------------------------------
// One-time conversion kernels
// ---------------------------------------------------------------------------
__global__ __launch_bounds__(256) void cvt_bf16_kernel(
    const float* __restrict__ in, unsigned short* __restrict__ out, long long n)
{
  const long long i = ((long long)blockIdx.x * blockDim.x + threadIdx.x) * 8;
  if (i >= n) return;
  float4 a = *(const float4*)(in + i);
  float4 b = *(const float4*)(in + i + 4);
  *(ushort8*)(out + i) = cvt8(a, b);
}

// out[Cn][R] = bf16( in[R][Cn] ^T ), tiles of 32x32
__global__ __launch_bounds__(256) void cvt_t_kernel(
    const float* __restrict__ in, unsigned short* __restrict__ out, int R, int Cn)
{
  __shared__ float tile[32][33];
  const int r0 = blockIdx.y * 32, c0 = blockIdx.x * 32;
  const int tr = threadIdx.x >> 5;   // 0..7
  const int tc = threadIdx.x & 31;
  #pragma unroll
  for (int i = 0; i < 32; i += 8)
    tile[tr + i][tc] = in[(long long)(r0 + tr + i) * Cn + c0 + tc];
  __syncthreads();
  #pragma unroll
  for (int i = 0; i < 32; i += 8)
    out[(long long)(c0 + tr + i) * R + r0 + tc] = f2bf(tile[tc][tr + i]);
}

// ---------------------------------------------------------------------------
// bf16 GEMM: C = alpha * (A[M,K] @ Bt[N,K]^T + bias)
//   OUT_MODE 0: f32 C[M][N]
//   OUT_MODE 1: bf16 C[M][N]
//   OUT_MODE 2: bf16 transposed: C[gm>>sbShift][N][gm & (SB-1)]
// Block tile 64x128, K-step 32, 8 waves, 4 WMMA tiles per wave.
// ---------------------------------------------------------------------------
template <int OUT_MODE>
__global__ __launch_bounds__(256) void gemm_bf16_kernel(
    const unsigned short* __restrict__ A, const unsigned short* __restrict__ Bt,
    const float* __restrict__ bias, void* __restrict__ C,
    int N, int K, float alpha, int bias_mode, int sbShift,
    long long sA_, long long sB_, long long sC_)
{
  __shared__ __attribute__((aligned(16))) unsigned short shA[64 * 40];
  __shared__ __attribute__((aligned(16))) unsigned short shB[128 * 40];

  const int bz = blockIdx.z;
  A  += (long long)bz * sA_;
  Bt += (long long)bz * sB_;

  const int m0 = blockIdx.x * 64;
  const int n0 = blockIdx.y * 128;
  const int t    = threadIdx.x;
  const int lane = t & 31;
  const int wave = t >> 5;
  const int half = lane >> 4;
  const int row  = lane & 15;
  const int mt   = wave & 3;          // M-tile 0..3
  const int ntb  = (wave >> 2) << 2;  // N-tile base: 0 or 4

  // A staging: chunk t -> row t>>2, 8-wide K group (t&3)*8 (perm'd position)
  const int am   = t >> 2;
  const int ac   = (t & 3) * 8;
  const int aPos = permA(ac);

  v8f acc[4] = {};

  for (int kt = 0; kt < K; kt += 32) {
    cp16(&shA[am * 40 + aPos], A + (long long)(m0 + am) * K + kt + ac);
    #pragma unroll
    for (int j = 0; j < 2; ++j) {
      const int idx = t + j * 256;
      const int bn = idx >> 2, bc = (idx & 3) * 8;
      cp16(&shB[bn * 40 + bc], Bt + (long long)(n0 + bn) * K + kt + bc);
    }
    cp_wait_barrier();

    v16bf af = ldfrag(&shA[(mt * 16 + row) * 40 + half * 16]);
    #pragma unroll
    for (int j = 0; j < 4; ++j) {
      v16bf bf = ldfrag(&shB[((ntb + j) * 16 + row) * 40 + half * 16]);
      acc[j] = wmma_bf16(af, bf, acc[j]);
    }
    __syncthreads();
  }

  // epilogue: D layout -> N = lane&15, M = r + 8*half
  float*          Cf = (float*)C          + (long long)bz * sC_;
  unsigned short* Ch = (unsigned short*)C + (long long)bz * sC_;
  #pragma unroll
  for (int j = 0; j < 4; ++j) {
    const int gn = n0 + (ntb + j) * 16 + row;
    const float bcol = (bias_mode == 1) ? bias[gn] : 0.f;
    #pragma unroll
    for (int r = 0; r < 8; ++r) {
      const int gm = m0 + mt * 16 + half * 8 + r;
      const float bval = (bias_mode == 2) ? bias[gm] : bcol;
      const float v = alpha * (acc[j][r] + bval);
      if constexpr (OUT_MODE == 0) {
        Cf[(long long)gm * N + gn] = v;
      } else if constexpr (OUT_MODE == 1) {
        Ch[(long long)gm * N + gn] = f2bf(v);
      } else {
        const int SB = 1 << sbShift;
        const int bb = gm >> sbShift;
        const int sm = gm & (SB - 1);
        Ch[((long long)bb * N + gn) * SB + sm] = f2bf(v);
      }
    }
  }
}

// ---------------------------------------------------------------------------
// Fused Linformer attention per (b, h, 32-row tile), all-bf16 inputs:
//   q    : bf16 [B][S][D]          (pre-scaled by 1/sqrt(Dk))
//   kproj: bf16 [B][Kp][D]
//   vpt  : bf16 [B][D][Kp]         (transposed V projection)
//   ctx  : bf16 [B][S][D]
// ---------------------------------------------------------------------------
__global__ __launch_bounds__(256) void linf_attn_kernel(
    const unsigned short* __restrict__ q, const unsigned short* __restrict__ kproj,
    const unsigned short* __restrict__ vpt, unsigned short* __restrict__ ctx)
{
  __shared__ __attribute__((aligned(16))) unsigned short sQ[32 * 72];   // A-perm
  __shared__ __attribute__((aligned(16))) unsigned short sK[256 * 72];  // B: [kp][d]
  __shared__ __attribute__((aligned(16))) unsigned short sV[64 * 264];  // B: [d][kp]
  __shared__ __attribute__((aligned(16))) unsigned short sP[32 * 264];  // A-perm probs
  __shared__ float sS[32 * 260];                                        // raw scores

  const int s0 = blockIdx.x * 32;
  const int h  = blockIdx.y;
  const int b  = blockIdx.z;
  const int t    = threadIdx.x;
  const int lane = t & 31;
  const int wave = t >> 5;
  const int half = lane >> 4;
  const int row  = lane & 15;

  // --- stage Q tile (32x64) in A-fragment order: one 16B chunk per thread ---
  {
    const int m = t >> 3, c = (t & 7) * 8;
    const int kb = c >> 5, kk = c & 31;
    cp16(&sQ[m * 72 + kb * 32 + permA(kk)],
         q + ((long long)b * S_LEN + s0 + m) * D_EMB + h * DK + c);
  }
  // --- stage K head slice (256x64), natural rows = B layout ---
  #pragma unroll
  for (int it = 0; it < 8; ++it) {
    const int idx = t + it * 256;
    const int n = idx >> 3, c = (idx & 7) * 8;
    cp16(&sK[n * 72 + c], kproj + ((long long)b * KP + n) * D_EMB + h * DK + c);
  }
  // --- stage V^T head slice: sV[d][kp] straight from vpt rows ---
  #pragma unroll
  for (int it = 0; it < 8; ++it) {
    const int idx = t + it * 256;
    const int d = idx >> 5, c = (idx & 31) * 8;
    cp16(&sV[d * 264 + c], vpt + ((long long)b * D_EMB + h * DK + d) * KP + c);
  }
  cp_wait_barrier();

  // --- scores: 2 M-tiles x 16 N-tiles, 4 per wave, K = 64 ---
  {
    const int mt  = wave & 1;
    const int ntb = (wave >> 1) << 2;
    v8f acc[4] = {};
    #pragma unroll
    for (int kb = 0; kb < 2; ++kb) {
      v16bf a = ldfrag(&sQ[(mt * 16 + row) * 72 + kb * 32 + half * 16]);
      #pragma unroll
      for (int j = 0; j < 4; ++j) {
        v16bf bb = ldfrag(&sK[((ntb + j) * 16 + row) * 72 + kb * 32 + half * 16]);
        acc[j] = wmma_bf16(a, bb, acc[j]);
      }
    }
    #pragma unroll
    for (int j = 0; j < 4; ++j)
      #pragma unroll
      for (int r = 0; r < 8; ++r)
        sS[(mt * 16 + half * 8 + r) * 260 + (ntb + j) * 16 + row] = acc[j][r];
  }
  __syncthreads();

  // --- row softmax, repacked into A-fragment order ---
  if (t < 32) {
    float mx = -3.0e38f;
    for (int i = 0; i < KP; ++i) mx = fmaxf(mx, sS[t * 260 + i]);
    float sum = 0.f;
    for (int i = 0; i < KP; ++i) {
      float e = __expf(sS[t * 260 + i] - mx);
      sS[t * 260 + i] = e;
      sum += e;
    }
    const float inv = 1.f / sum;
    for (int i = 0; i < KP; ++i) {
      const int kb = i >> 5, kk = i & 31;
      sP[t * 264 + kb * 32 + permA(kk)] = f2bf(sS[t * 260 + i] * inv);
    }
  }
  __syncthreads();

  // --- ctx = P(32x256) @ V(256x64): one 16x16 tile per wave ---
  {
    const int mt = wave & 1;
    const int nt = wave >> 1;
    v8f acc = {};
    #pragma unroll
    for (int kb = 0; kb < 8; ++kb) {
      v16bf a  = ldfrag(&sP[(mt * 16 + row) * 264 + kb * 32 + half * 16]);
      v16bf bb = ldfrag(&sV[(nt * 16 + row) * 264 + kb * 32 + half * 16]);
      acc = wmma_bf16(a, bb, acc);
    }
    #pragma unroll
    for (int r = 0; r < 8; ++r) {
      const int gm = s0 + mt * 16 + half * 8 + r;
      const int gd = h * DK + nt * 16 + row;
      ctx[((long long)b * S_LEN + gm) * D_EMB + gd] = f2bf(acc[r]);
    }
  }
}

// ---------------------------------------------------------------------------
// Host launcher
// ---------------------------------------------------------------------------
extern "C" void kernel_launch(void* const* d_in, const int* in_sizes, int n_in,
                              void* d_out, int out_size, void* d_ws, size_t ws_size,
                              hipStream_t stream) {
  (void)in_sizes; (void)n_in; (void)out_size; (void)ws_size;
  const float* query = (const float*)d_in[0];
  const float* value = (const float*)d_in[1];
  const float* Wq = (const float*)d_in[2];
  const float* bq = (const float*)d_in[3];
  const float* Wk = (const float*)d_in[4];
  const float* bk = (const float*)d_in[5];
  const float* Wv = (const float*)d_in[6];
  const float* bv = (const float*)d_in[7];
  const float* Wo = (const float*)d_in[8];
  const float* bo = (const float*)d_in[9];
  const float* E  = (const float*)d_in[10];
  const float* Eb = (const float*)d_in[11];
  const float* F  = (const float*)d_in[12];
  const float* Fb = (const float*)d_in[13];
  float* out = (float*)d_out;

  const int Bn = 4, S = S_LEN, D = D_EMB, H = 16, Kp = KP;
  const int BS = Bn * S;                      // 8192
  const long long BSD = (long long)BS * D;    // 8,388,608

  unsigned short* us = (unsigned short*)d_ws;
  long long offs = 0;
  auto alloc = [&](long long n) { unsigned short* p = us + offs; offs += n; return p; };
  unsigned short* qf  = alloc(BSD);                  // bf16(query)
  unsigned short* vf  = alloc(BSD);                  // bf16(value)
  unsigned short* Wqt = alloc((long long)D * D);     // Wq^T
  unsigned short* Wkt = alloc((long long)D * D);
  unsigned short* Wvt = alloc((long long)D * D);
  unsigned short* Wot = alloc((long long)D * D);
  unsigned short* Et  = alloc((long long)Kp * S);    // E^T [Kp][S]
  unsigned short* Ft  = alloc((long long)Kp * S);
  unsigned short* qB  = alloc(BSD);                  // q bf16 [BS][D]
  unsigned short* kT  = alloc(BSD);                  // k^T  [Bn][D][S]
  unsigned short* vT  = alloc(BSD);                  // v^T  [Bn][D][S]
  unsigned short* kpj = alloc((long long)Bn * Kp * D); // kproj [Bn][Kp][D]
  unsigned short* vpT = alloc((long long)Bn * Kp * D); // vproj^T [Bn][D][Kp]
  unsigned short* ctx = alloc(BSD);                  // ctx bf16 [BS][D]

  const dim3 blk(256);

  // one-time conversions / transposes
  cvt_bf16_kernel<<<dim3((unsigned)(BSD / 2048)), blk, 0, stream>>>(query, qf, BSD);
  cvt_bf16_kernel<<<dim3((unsigned)(BSD / 2048)), blk, 0, stream>>>(value, vf, BSD);
  cvt_t_kernel<<<dim3(D / 32, D / 32), blk, 0, stream>>>(Wq, Wqt, D, D);
  cvt_t_kernel<<<dim3(D / 32, D / 32), blk, 0, stream>>>(Wk, Wkt, D, D);
  cvt_t_kernel<<<dim3(D / 32, D / 32), blk, 0, stream>>>(Wv, Wvt, D, D);
  cvt_t_kernel<<<dim3(D / 32, D / 32), blk, 0, stream>>>(Wo, Wot, D, D);
  cvt_t_kernel<<<dim3(Kp / 32, S / 32), blk, 0, stream>>>(E, Et, S, Kp);
  cvt_t_kernel<<<dim3(Kp / 32, S / 32), blk, 0, stream>>>(F, Ft, S, Kp);

  // q = (1/8)*(query@Wq + bq) -> bf16 [BS][D]
  gemm_bf16_kernel<1><<<dim3(BS / 64, D / 128, 1), blk, 0, stream>>>(
      qf, Wqt, bq, qB, D, D, 0.125f, 1, 0, 0, 0, 0);
  // k^T, v^T = (value@W + b)^T per batch -> [Bn][D][S]  (sbShift=11: SB=2048)
  gemm_bf16_kernel<2><<<dim3(BS / 64, D / 128, 1), blk, 0, stream>>>(
      vf, Wkt, bk, kT, D, D, 1.0f, 1, 11, 0, 0, 0);
  gemm_bf16_kernel<2><<<dim3(BS / 64, D / 128, 1), blk, 0, stream>>>(
      vf, Wvt, bv, vT, D, D, 1.0f, 1, 11, 0, 0, 0);

  // kproj[b] = E^T @ k[b] + Eb(row) -> bf16 [Kp][D]   (A=Et, Bt=kT[b])
  gemm_bf16_kernel<1><<<dim3(Kp / 64, D / 128, Bn), blk, 0, stream>>>(
      Et, kT, Eb, kpj, D, S, 1.0f, 2, 0,
      0, (long long)D * S, (long long)Kp * D);
  // vproj[b]^T = (F^T @ v[b] + Fb(row))^T -> bf16 [D][Kp]  (sbShift=8: SB=256)
  gemm_bf16_kernel<2><<<dim3(Kp / 64, D / 128, Bn), blk, 0, stream>>>(
      Ft, vT, Fb, vpT, D, S, 1.0f, 2, 8,
      0, (long long)D * S, (long long)Kp * D);

  // fused attention -> ctx bf16
  linf_attn_kernel<<<dim3(S / 32, H, Bn), blk, 0, stream>>>(qB, kpj, vpT, ctx);

  // out = ctx@Wo + bo -> f32 d_out
  gemm_bf16_kernel<0><<<dim3(BS / 64, D / 128, 1), blk, 0, stream>>>(
      ctx, Wot, bo, out, D, D, 1.0f, 1, 0, 0, 0, 0);
}
